// node2grid_encoder_62577673503495
// MI455X (gfx1250) — matrix operand order repaired and checked by reading, not compiled, not run on hardware
//
#include <hip/hip_runtime.h>
#include <hip/hip_bf16.h>

// CDNA5 / gfx1250: wave32, WMMA 16x16x32 bf16 -> f32 accumulate.
typedef __attribute__((ext_vector_type(16))) __bf16 v16bf;
typedef __attribute__((ext_vector_type(8)))  __bf16 v8bf;
typedef __attribute__((ext_vector_type(4)))  __bf16 v4bf;
typedef __attribute__((ext_vector_type(8)))  float  v8f;

#define BT   384   // B*T = 8*48
#define NN   1024  // nodes
#define DD   64    // feature dim
#define GG   256   // grid cells

// ---------------------------------------------------------------------------
// Kernel 0: one-time view f32 -> bf16 (1 MB -> 512 KB, L2-resident afterwards).
// ---------------------------------------------------------------------------
__global__ __launch_bounds__(256) void n2g_view_cvt(
    const float* __restrict__ vin, __bf16* __restrict__ vout)
{
  int idx = blockIdx.x * 256 + threadIdx.x;   // float4 index, GG*NN/4 total
  float4 f = ((const float4*)vin)[idx];
  v4bf p;
  p[0] = (__bf16)f.x; p[1] = (__bf16)f.y;
  p[2] = (__bf16)f.z; p[3] = (__bf16)f.w;
  ((v4bf*)vout)[idx] = p;
}

// ---------------------------------------------------------------------------
// Kernel 1: h[bt][n][d] = x[bt][n][:] @ w[n] + b[n][d], stored as bf16.
// One block = one node n, 128 bt rows (8 waves x 16-row tiles).
// ---------------------------------------------------------------------------
__global__ __launch_bounds__(256) void n2g_node_xform(
    const float* __restrict__ x,      // [BT][NN][DD]
    const float* __restrict__ w,      // [NN][DD][DD]  (i-major: w[n][i][o])
    const float* __restrict__ bias,   // [NN][DD]
    __bf16* __restrict__ h)           // [BT][NN][DD] bf16
{
  __shared__ __align__(16) __bf16 wT[DD * DD];        // transposed: wT[o][i], 8 KB
  __shared__ __align__(16) __bf16 stage[8][16 * DD];  // per-wave 16x64 tile, 16 KB

  const int n    = blockIdx.x;
  const int tid  = threadIdx.x;
  const int lane = tid & 31;
  const int wave = tid >> 5;
  const int m    = lane & 15;
  const int half = lane >> 4;

  // --- load w[n] (f32, row-major i x o) -> bf16 transposed wT[o][i] ---
  {
    const float4* wsrc = (const float4*)(w + (size_t)n * DD * DD);
#pragma unroll
    for (int c = 0; c < 4; ++c) {
      int idx = tid * 4 + c;          // 1024 float4 chunks total
      int i   = idx >> 4;             // input dim (K)
      int o4  = (idx & 15) << 2;      // output col base
      float4 f = wsrc[idx];
      wT[(o4 + 0) * DD + i] = (__bf16)f.x;
      wT[(o4 + 1) * DD + i] = (__bf16)f.y;
      wT[(o4 + 2) * DD + i] = (__bf16)f.z;
      wT[(o4 + 3) * DD + i] = (__bf16)f.w;
    }
  }
  __syncthreads();

  const int rowbase = blockIdx.y * 128 + wave * 16;
  const int btrow   = rowbase + m;
  const float* xrow = x + ((size_t)btrow * NN + n) * DD;

  v8f acc[4] = {};  // 4 d-tiles of 16 cols, 16 bt rows

#pragma unroll
  for (int ks = 0; ks < 2; ++ks) {    // K = 64 = 2 x 32
    // A fragment: 16x32 bf16, contiguous K order (half selects K 0-15 vs 16-31)
    v16bf a;
#pragma unroll
    for (int i = 0; i < 8; ++i) {
      int k = ks * 32 + (half << 4) + (i << 1);
      float2 f = *(const float2*)(xrow + k);
      a[2 * i]     = (__bf16)f.x;
      a[2 * i + 1] = (__bf16)f.y;
    }
    // batch all 4 B fragments, then 4 back-to-back WMMAs
    v16bf bm[4];
#pragma unroll
    for (int t = 0; t < 4; ++t)
      bm[t] = *(const v16bf*)(wT + (t * 16 + m) * DD + ks * 32 + half * 16);
#pragma unroll
    for (int t = 0; t < 4; ++t)
      acc[t] = __builtin_amdgcn_wmma_f32_16x16x32_bf16(
          false, a, false, bm[t], (short)0, acc[t], false, false);
  }

  // bias add (per output column d, broadcast over rows)
#pragma unroll
  for (int t = 0; t < 4; ++t) {
    float bv = bias[n * DD + t * 16 + m];
#pragma unroll
    for (int r = 0; r < 8; ++r) acc[t][r] += bv;
  }

  // stage bf16 tile in LDS (C layout: lane col = m, row = r + 8*half)
  __bf16* st = stage[wave];
#pragma unroll
  for (int t = 0; t < 4; ++t)
#pragma unroll
    for (int r = 0; r < 8; ++r)
      st[(r + half * 8) * DD + t * 16 + m] = (__bf16)acc[t][r];

  // cooperative coalesced writeout: lane -> half a 128B row (4 x b128)
  {
    int mr = lane >> 1;
    int el = (lane & 1) * 32;  // 32 bf16 = 64B
    const uint4* s4 = (const uint4*)(st + mr * DD + el);
    uint4* d4 = (uint4*)(h + ((size_t)(rowbase + mr) * NN + n) * DD + el);
    d4[0] = s4[0]; d4[1] = s4[1]; d4[2] = s4[2]; d4[3] = s4[3];
  }
}

// ---------------------------------------------------------------------------
// Kernel 2 helpers
// ---------------------------------------------------------------------------
struct B4 { v16bf t[4]; };

__device__ __forceinline__ B4 load_b4(const __bf16* hT, int m, int half, int ks) {
  B4 b;
#pragma unroll
  for (int t = 0; t < 4; ++t)
    b.t[t] = *(const v16bf*)(hT + (size_t)(t * 16 + m) * NN + ks * 32 + half * 16);
  return b;
}

__device__ __forceinline__ v16bf load_a(const __bf16* vrow, int ks, int half) {
  // Contiguous A layout: lane-half h holds K 0-15 (h=0) vs 16-31 (h=1):
  // the fragment is 16 consecutive bf16 in memory.
  v8bf lo = *(const v8bf*)(vrow + ks * 32 + half * 16);
  v8bf hi = *(const v8bf*)(vrow + ks * 32 + half * 16 + 8);
  v16bf a;
#pragma unroll
  for (int j = 0; j < 8; ++j) { a[j] = lo[j]; a[8 + j] = hi[j]; }
  return a;
}

__device__ __forceinline__ void mm8(v8f (&acc)[2][4], const v16bf& a0,
                                    const v16bf& a1, const B4& b) {
#pragma unroll
  for (int t = 0; t < 4; ++t)
    acc[0][t] = __builtin_amdgcn_wmma_f32_16x16x32_bf16(
        false, a0, false, b.t[t], (short)0, acc[0][t], false, false);
#pragma unroll
  for (int t = 0; t < 4; ++t)
    acc[1][t] = __builtin_amdgcn_wmma_f32_16x16x32_bf16(
        false, a1, false, b.t[t], (short)0, acc[1][t], false, false);
}

__device__ __forceinline__ unsigned getc(const uint4& v, int c) {
  return c == 0 ? v.x : c == 1 ? v.y : c == 2 ? v.z : v.w;
}

// ---------------------------------------------------------------------------
// Kernel 2: g[bt][g][:] = view[g][:] @ h[bt][:][:]  then LayerNorm over D.
// One block = one bt, all 256 g rows; each wave owns two 16-row tiles so the
// 4 B fragments per k-step feed 8 WMMAs. B fragments ping-pong (no reg copies).
// ---------------------------------------------------------------------------
__global__ __launch_bounds__(256) void n2g_grid_agg_ln(
    const __bf16* __restrict__ viewb, // [GG][NN] bf16
    const __bf16* __restrict__ h,     // [BT][NN][DD] bf16
    const float* __restrict__ gamma,  // [DD]
    const float* __restrict__ beta,   // [DD]
    float* __restrict__ out)          // [BT][GG][DD]
{
  extern __shared__ __align__(16) __bf16 hT[];  // [DD][NN] = 128 KB

  const int bt   = blockIdx.x;
  const int tid  = threadIdx.x;
  const int lane = tid & 31;
  const int wave = tid >> 5;
  const int m    = lane & 15;
  const int half = lane >> 4;

  // --- stage h_bt [n][d] -> hT[d][n]; v_perm packs two bf16 per instruction ---
  {
    const __bf16* hsrc = h + (size_t)bt * NN * DD;
    int n0 = (tid >> 1) * 8;        // 8-row n group
    int jb = (tid & 1) * 4;         // which half of the 8 uint4 d-chunks
#pragma unroll
    for (int jj = 0; jj < 4; ++jj) {
      int j = jb + jj;              // d = j*8 .. j*8+7
      uint4 r[8];
#pragma unroll
      for (int i = 0; i < 8; ++i)
        r[i] = ((const uint4*)(hsrc + (size_t)(n0 + i) * DD))[j];
#pragma unroll
      for (int dd = 0; dd < 8; ++dd) {
        const int      c   = dd >> 1;
        const unsigned sel = (dd & 1) ? 0x07060302u : 0x05040100u;
        uint4 p;
        p.x = __builtin_amdgcn_perm(getc(r[1], c), getc(r[0], c), sel);
        p.y = __builtin_amdgcn_perm(getc(r[3], c), getc(r[2], c), sel);
        p.z = __builtin_amdgcn_perm(getc(r[5], c), getc(r[4], c), sel);
        p.w = __builtin_amdgcn_perm(getc(r[7], c), getc(r[6], c), sel);
        *(uint4*)((unsigned short*)hT + (size_t)(j * 8 + dd) * NN + n0) = p;
      }
    }
  }
  __syncthreads();

  const int gt0 = wave * 32;                // two 16-row tiles per wave
  const __bf16* vb0 = viewb + (size_t)(gt0 + m) * NN;
  const __bf16* vb1 = viewb + (size_t)(gt0 + 16 + m) * NN;

  v8f acc[2][4] = {};  // [row-tile][d-tile]: 32 g rows x 64 d cols

  // ping-pong B buffers across an unroll-2 k-loop: no register rotation moves
  B4 bcur = load_b4(hT, m, half, 0);
  for (int ks = 0; ks < 32; ks += 2) {   // K = 1024 = 32 x 32
    __builtin_prefetch(vb0 + ((ks * 32 + 256) & (NN - 1)), 0, 1);

    B4 bodd = load_b4(hT, m, half, ks + 1);
    v16bf a0 = load_a(vb0, ks, half);
    v16bf a1 = load_a(vb1, ks, half);
    mm8(acc, a0, a1, bcur);

    bcur = load_b4(hT, m, half, (ks + 2) & 31);  // wrap: harmless last-iter load
    a0 = load_a(vb0, ks + 1, half);
    a1 = load_a(vb1, ks + 1, half);
    mm8(acc, a0, a1, bodd);
  }

  // --- fused LayerNorm: each g-row's 64 d-values live in one 16-lane half ---
#pragma unroll
  for (int rt = 0; rt < 2; ++rt) {
    float mu[8], rsd[8];
#pragma unroll
    for (int r = 0; r < 8; ++r) {
      float s = acc[rt][0][r] + acc[rt][1][r] + acc[rt][2][r] + acc[rt][3][r];
      float q = acc[rt][0][r] * acc[rt][0][r] + acc[rt][1][r] * acc[rt][1][r] +
                acc[rt][2][r] * acc[rt][2][r] + acc[rt][3][r] * acc[rt][3][r];
#pragma unroll
      for (int off = 1; off < 16; off <<= 1) {
        s += __shfl_xor(s, off, 32);
        q += __shfl_xor(q, off, 32);
      }
      float mean = s * (1.0f / 64.0f);
      float var  = q * (1.0f / 64.0f) - mean * mean;
      mu[r]  = mean;
      rsd[r] = rsqrtf(var + 1e-5f);
    }
#pragma unroll
    for (int t = 0; t < 4; ++t) {
      int d = t * 16 + m;
      float ga = gamma[d], be = beta[d];
#pragma unroll
      for (int r = 0; r < 8; ++r) {
        int grow = gt0 + rt * 16 + r + half * 8;
        out[((size_t)bt * GG + grow) * DD + d] =
            (acc[rt][t][r] - mu[r]) * rsd[r] * ga + be;
      }
    }
  }
}

// ---------------------------------------------------------------------------
extern "C" void kernel_launch(void* const* d_in, const int* in_sizes, int n_in,
                              void* d_out, int out_size, void* d_ws, size_t ws_size,
                              hipStream_t stream) {
  const float* x     = (const float*)d_in[0];  // [8,48,1024,64]
  const float* view  = (const float*)d_in[1];  // [256,1024]
  const float* w     = (const float*)d_in[2];  // [1024,64,64]
  const float* bias  = (const float*)d_in[3];  // [1,1024,64]
  const float* gamma = (const float*)d_in[4];  // [64]
  const float* beta  = (const float*)d_in[5];  // [64]
  float* out = (float*)d_out;                  // [8,48,256,64]

  // workspace: h (bf16, 48 MiB) then view in bf16 (512 KiB)
  __bf16* hbuf  = (__bf16*)d_ws;
  __bf16* viewb = (__bf16*)((char*)d_ws + (size_t)BT * NN * DD * sizeof(__bf16));

  n2g_view_cvt<<<dim3(GG * NN / 4 / 256), 256, 0, stream>>>(view, viewb);

  n2g_node_xform<<<dim3(NN, BT / 128), 256, 0, stream>>>(x, w, bias, hbuf);

  const size_t lds2 = (size_t)DD * NN * sizeof(__bf16);  // 128 KB
  n2g_grid_agg_ln<<<dim3(BT), 256, lds2, stream>>>(viewb, hbuf, gamma, beta, out);
}
